// IseerBlock_76012331204740
// MI455X (gfx1250) — compile-verified, matrix operand order checked by reading, not run on hardware
//
#include <hip/hip_runtime.h>
#include <hip/hip_bf16.h>

// ---------------- problem constants ----------------
#define Bq     2
#define Lq     2048
#define Dq     1024
#define DI     2048          // D_INNER
#define DSTATE 16
#define DTRANK 64
#define TT     (Bq*Lq)       // 4096 tokens
#define NE     8
#define NPROJ  96            // DT_RANK + 2*D_STATE

typedef __attribute__((ext_vector_type(16))) __bf16 v16bf;
typedef __attribute__((ext_vector_type(8)))  __bf16 v8bf;
typedef __attribute__((ext_vector_type(8)))  float  v8f;

union BFrag { v16bf v; v8bf h[2]; };

// LDS byte offset of a shared-memory location: per CDNA5 aperture rules the
// flat address of an LDS location carries the LDS offset in its low 32 bits.
__device__ __forceinline__ unsigned lds_off(const void* p)
{
    return (unsigned)(unsigned long long)(uintptr_t)p;
}

// Issue one per-lane 16B global -> LDS async copy (tracked by ASYNCcnt).
__device__ __forceinline__ void async_copy_b128(unsigned lds, const void* gptr)
{
    asm volatile("global_load_async_to_lds_b128 %0, %1, off"
                 :: "v"(lds), "v"((unsigned long long)(uintptr_t)gptr)
                 : "memory");
}

__device__ __forceinline__ void wait_async0()
{
    asm volatile("s_wait_asynccnt 0x0" ::: "memory");
}

// =====================================================================
// bf16 WMMA GEMM: C(MxN) = A(MxK) * B(KxN) with B given PRE-TRANSPOSED
// (Bt is N x K row-major bf16).  Block = 256 thr (8 wave32), tile 128x128,
// each wave 32x64 (2x4 WMMA accumulators).  Double-buffered LDS staging via
// global_load_async_to_lds_b128, prefetching K-step s+1 under step s.
// modes: 0: C=acc    1: C=resid+acc    2: Cbf=bf16(gelu(acc+bias))
//        3: C += gates[t,expert]*(acc+bias)
// =====================================================================
__global__ __launch_bounds__(256)
void gemm_bf16_wmma(const __bf16* __restrict__ A, const __bf16* __restrict__ Bt,
                    int M, int N, int K,
                    float* __restrict__ C, const float* __restrict__ resid,
                    const float* __restrict__ bias, const float* __restrict__ gates,
                    int expert, __bf16* __restrict__ Cbf, int mode)
{
    __shared__ __bf16 sA[2][128 * 32];   // A tile  (m, k) row-major
    __shared__ __bf16 sB[2][128 * 32];   // Bt tile (n, k) row-major

    const int tid  = threadIdx.x;
    const int wave = tid >> 5, lane = tid & 31;
    const int wm   = wave & 3;           // 0..3 -> 32-row M slice
    const int wn   = wave >> 2;          // 0..1 -> 64-col N slice
    const int m16  = lane & 15;
    const int ksel = lane >> 4;          // lanes 16-31 hold the +8 / +16 K half
    const int bm   = blockIdx.y * 128, bn = blockIdx.x * 128;

    v8f acc[2][4];
    #pragma unroll
    for (int i = 0; i < 2; ++i)
        #pragma unroll
        for (int j = 0; j < 4; ++j)
            #pragma unroll
            for (int r = 0; r < 8; ++r) acc[i][j][r] = 0.f;

    // Per thread: 2 chunks of A + 2 chunks of Bt per stage (512 x 16B each).
    const int c0   = tid * 2;
    const int row0 = c0 >> 2,       cc0 = (c0 & 3) * 8;
    const int row1 = (c0 + 1) >> 2, cc1 = ((c0 + 1) & 3) * 8;

    auto stage = [&](int buf, int k0) {
        async_copy_b128(lds_off(&sA[buf][row0 * 32 + cc0]),
                        A + (size_t)(bm + row0) * K + k0 + cc0);
        async_copy_b128(lds_off(&sA[buf][row1 * 32 + cc1]),
                        A + (size_t)(bm + row1) * K + k0 + cc1);
        async_copy_b128(lds_off(&sB[buf][row0 * 32 + cc0]),
                        Bt + (size_t)(bn + row0) * K + k0 + cc0);
        async_copy_b128(lds_off(&sB[buf][row1 * 32 + cc1]),
                        Bt + (size_t)(bn + row1) * K + k0 + cc1);
    };

    const int nsteps = K >> 5;
    stage(0, 0);

    for (int s = 0; s < nsteps; ++s) {
        // Stage-s data resident; all waves done reading buffer s^1.
        wait_async0();
        __syncthreads();
        if (s + 1 < nsteps) stage((s + 1) & 1, (s + 1) << 5);   // prefetch

        const __bf16* a = sA[s & 1];
        const __bf16* b = sB[s & 1];
        BFrag af[2], bf[4];
        const int kb = ksel * 8;
        #pragma unroll
        for (int i = 0; i < 2; ++i) {
            int row = wm * 32 + i * 16 + m16;
            af[i].h[0] = *(const v8bf*)(&a[row * 32 + kb]);        // K = kb..kb+7
            af[i].h[1] = *(const v8bf*)(&a[row * 32 + 16 + kb]);   // K = 16+kb..
        }
        #pragma unroll
        for (int j = 0; j < 4; ++j) {
            int col = wn * 64 + j * 16 + m16;
            bf[j].h[0] = *(const v8bf*)(&b[col * 32 + kb]);
            bf[j].h[1] = *(const v8bf*)(&b[col * 32 + 16 + kb]);
        }
        #pragma unroll
        for (int i = 0; i < 2; ++i)
            #pragma unroll
            for (int j = 0; j < 4; ++j)
                acc[i][j] = __builtin_amdgcn_wmma_f32_16x16x32_bf16(
                    false, af[i].v, false, bf[j].v, (short)0, acc[i][j],
                    false, false);
        __syncthreads();   // readers done before buffer s&1 is overwritten
    }

    // ---- epilogue: VGPR r of the 16x16 f32 C-tile is row (ksel*8 + r), col m16 ----
    #pragma unroll
    for (int i = 0; i < 2; ++i) {
        #pragma unroll
        for (int j = 0; j < 4; ++j) {
            int gmb = bm + wm * 32 + i * 16 + ksel * 8;
            int gn  = bn + wn * 64 + j * 16 + m16;
            #pragma unroll
            for (int r = 0; r < 8; ++r) {
                int gm = gmb + r;
                float v = acc[i][j][r];
                size_t o = (size_t)gm * N + gn;
                if (mode == 0) {
                    C[o] = v;
                } else if (mode == 1) {
                    C[o] = resid[o] + v;
                } else if (mode == 2) {
                    float u = v + bias[gn];
                    float g = 0.5f * u *
                        (1.f + tanhf(0.7978845608f * (u + 0.044715f * u * u * u)));
                    Cbf[o] = (__bf16)g;
                } else {
                    float gt = gates[(size_t)gm * NE + expert];
                    C[o] += gt * (v + bias[gn]);
                }
            }
        }
    }
}

// ---------------- weight convert+transpose: f32 (K x N) -> bf16 (N x K) ----------------
// 32x32 LDS-tiled so both the global read and the global write are coalesced.
__global__ __launch_bounds__(256)
void wtrans_k(const float* __restrict__ in, __bf16* __restrict__ out, int K, int N)
{
    __shared__ float tile[32][33];
    const int tx = threadIdx.x & 31, ty = threadIdx.x >> 5;   // 32 x 8
    const int kb = blockIdx.y * 32, nb = blockIdx.x * 32;
    #pragma unroll
    for (int r = ty; r < 32; r += 8)
        tile[r][tx] = in[(size_t)(kb + r) * N + nb + tx];
    __syncthreads();
    #pragma unroll
    for (int r = ty; r < 32; r += 8)
        out[(size_t)(nb + r) * K + kb + tx] = (__bf16)tile[tx][r];
}

// ---------------- RMSNorm (1 block / token), f32 in -> bf16 (+opt f32) ----------------
__global__ __launch_bounds__(256)
void rmsnorm_k(const float* __restrict__ x, const float* __restrict__ w,
               __bf16* __restrict__ obf, float* __restrict__ of)
{
    const int t = blockIdx.x, tid = threadIdx.x;
    const int lane = tid & 31, wave = tid >> 5;
    __shared__ float red[8];
    float ss = 0.f;
    for (int i = tid; i < Dq; i += 256) {
        float v = x[(size_t)t * Dq + i];
        ss += v * v;
    }
    #pragma unroll
    for (int o = 16; o > 0; o >>= 1) ss += __shfl_xor(ss, o, 32);
    if (lane == 0) red[wave] = ss;
    __syncthreads();
    if (tid == 0) {
        float s = 0.f;
        for (int i = 0; i < 8; ++i) s += red[i];
        red[0] = rsqrtf(s / (float)Dq + 1e-6f);
    }
    __syncthreads();
    float rinv = red[0];
    for (int i = tid; i < Dq; i += 256) {
        float v = x[(size_t)t * Dq + i] * rinv * w[i];
        obf[(size_t)t * Dq + i] = (__bf16)v;
        if (of) of[(size_t)t * Dq + i] = v;
    }
}

// ---------------- causal depthwise conv1d + SiLU ----------------
__global__ void conv_silu_k(const float* __restrict__ xz, const float* __restrict__ cw,
                            const float* __restrict__ cb, float* __restrict__ xcs)
{
    int idx = blockIdx.x * 256 + threadIdx.x;   // < TT*DI
    int di = idx & (DI - 1);
    int t  = idx >> 11;
    int l  = t & (Lq - 1);
    int b  = t >> 11;
    float acc = cb[di];
    #pragma unroll
    for (int k = 0; k < 4; ++k) {
        int ls = l - 3 + k;
        if (ls >= 0)
            acc += xz[(size_t)(b * Lq + ls) * (2 * DI) + di] * cw[di * 4 + k];
    }
    acc = acc / (1.f + __expf(-acc));
    xcs[(size_t)t * DI + di] = acc;
}

// ---------------- x_proj: (T,2048) @ (2048,96) -- small-N VALU GEMM ----------------
__global__ void xproj_k(const float* __restrict__ xcs, const float* __restrict__ w,
                        float* __restrict__ proj)
{
    int idx = blockIdx.x * 256 + threadIdx.x;   // < TT*96
    int t = idx / NPROJ, n = idx % NPROJ;
    const float* xr = xcs + (size_t)t * DI;
    float acc = 0.f;
    for (int k = 0; k < DI; ++k) acc += xr[k] * w[k * NPROJ + n];
    proj[idx] = acc;
}

// ---------------- dt = softplus(proj[:, :64] @ dt_proj_w + b) ----------------
__global__ void dt_k(const float* __restrict__ proj, const float* __restrict__ w,
                     const float* __restrict__ b, float* __restrict__ dts)
{
    int idx = blockIdx.x * 256 + threadIdx.x;   // < TT*DI
    int t = idx >> 11, di = idx & (DI - 1);
    float acc = b[di];
    const float* pr = proj + (size_t)t * NPROJ;
    #pragma unroll 8
    for (int r = 0; r < DTRANK; ++r) acc += pr[r] * w[r * DI + di];
    dts[idx] = (acc > 20.f) ? acc : log1pf(__expf(acc));
}

// ---------------- selective scan: one thread per (b, d_inner) channel ----------------
// Fuses the D*x skip and the SiLU(z) output gate; emits bf16 for out_proj WMMA.
__global__ __launch_bounds__(256)
void scan_k(const float* __restrict__ xcs, const float* __restrict__ dts,
            const float* __restrict__ proj, const float* __restrict__ xz,
            const float* __restrict__ A_log, const float* __restrict__ Dp,
            __bf16* __restrict__ ybf)
{
    int idx = blockIdx.x * 256 + threadIdx.x;   // < Bq*DI = 4096
    int di = idx & (DI - 1), b = idx >> 11;
    float a[DSTATE], st[DSTATE];
    #pragma unroll
    for (int s = 0; s < DSTATE; ++s) {
        a[s]  = -__expf(A_log[di * DSTATE + s]);
        st[s] = 0.f;
    }
    float dskip = Dp[di];
    for (int l = 0; l < Lq; ++l) {
        size_t t = (size_t)b * Lq + l;
        float xv  = xcs[t * DI + di];
        float dtv = dts[t * DI + di];
        const float* pr = proj + t * NPROJ;
        float y = 0.f, dx = dtv * xv;
        #pragma unroll
        for (int s = 0; s < DSTATE; ++s) {
            float dA = __expf(dtv * a[s]);
            st[s] = st[s] * dA + dx * pr[DTRANK + s];          // B_t
            y += st[s] * pr[DTRANK + DSTATE + s];              // C_t
        }
        y += dskip * xv;
        float z = xz[t * (2 * DI) + DI + di];
        y *= z / (1.f + __expf(-z));
        ybf[t * DI + di] = (__bf16)y;
    }
}

// ---------------- router: 1 wave32 / token; softmax + top2 + aux stats ----------------
__global__ __launch_bounds__(256)
void router_k(const float* __restrict__ ht, const float* __restrict__ rw,
              float* __restrict__ gates, float* __restrict__ psum,
              float* __restrict__ dsum)
{
    int wave = threadIdx.x >> 5, lane = threadIdx.x & 31;
    int t = blockIdx.x * 8 + wave;
    float lg[NE];
    #pragma unroll
    for (int e = 0; e < NE; ++e) lg[e] = 0.f;
    for (int k = lane; k < Dq; k += 32) {
        float h = ht[(size_t)t * Dq + k];
        #pragma unroll
        for (int e = 0; e < NE; ++e) lg[e] += h * rw[k * NE + e];
    }
    #pragma unroll
    for (int e = 0; e < NE; ++e)
        #pragma unroll
        for (int o = 16; o > 0; o >>= 1) lg[e] += __shfl_xor(lg[e], o, 32);
    if (lane == 0) {
        float m = lg[0];
        for (int e = 1; e < NE; ++e) m = fmaxf(m, lg[e]);
        float p[NE], s = 0.f;
        for (int e = 0; e < NE; ++e) { p[e] = __expf(lg[e] - m); s += p[e]; }
        for (int e = 0; e < NE; ++e) p[e] /= s;
        int i0 = 0;
        for (int e = 1; e < NE; ++e) if (p[e] > p[i0]) i0 = e;
        int i1 = (i0 == 0) ? 1 : 0;
        for (int e = 0; e < NE; ++e) if (e != i0 && p[e] > p[i1]) i1 = e;
        float g = p[i0] + p[i1];
        for (int e = 0; e < NE; ++e)
            gates[(size_t)t * NE + e] =
                (e == i0) ? p[i0] / g : (e == i1) ? p[i1] / g : 0.f;
        for (int e = 0; e < NE; ++e) atomicAdd(&psum[e], p[e]);
        atomicAdd(&dsum[i0], 1.f);
        atomicAdd(&dsum[i1], 1.f);
    }
}

__global__ void zero32_k(float* p) { if (threadIdx.x < 32) p[threadIdx.x] = 0.f; }

__global__ void aux_k(const float* __restrict__ psum, const float* __restrict__ dsum,
                      float* __restrict__ out)
{
    if (threadIdx.x == 0 && blockIdx.x == 0) {
        float s = 0.f;
        for (int e = 0; e < NE; ++e)
            s += (psum[e] / (float)TT) * (dsum[e] / (float)TT);
        out[0] = (float)NE * s / 2.0f;   // E * sum / TOPK
    }
}

// =====================================================================
extern "C" void kernel_launch(void* const* d_in, const int* in_sizes, int n_in,
                              void* d_out, int out_size, void* d_ws, size_t ws_size,
                              hipStream_t stream)
{
    (void)in_sizes; (void)n_in; (void)out_size; (void)ws_size;
    const float* x     = (const float*)d_in[0];
    const float* n1w   = (const float*)d_in[1];
    const float* n2w   = (const float*)d_in[2];
    const float* inpw  = (const float*)d_in[3];
    const float* convw = (const float*)d_in[4];
    const float* convb = (const float*)d_in[5];
    const float* xpw   = (const float*)d_in[6];
    const float* dtw   = (const float*)d_in[7];
    const float* dtb   = (const float*)d_in[8];
    const float* alog  = (const float*)d_in[9];
    const float* dpar  = (const float*)d_in[10];
    const float* outw  = (const float*)d_in[11];
    const float* rw    = (const float*)d_in[12];
    const float* w1    = (const float*)d_in[13];
    const float* b1    = (const float*)d_in[14];
    const float* w2    = (const float*)d_in[15];
    const float* b2    = (const float*)d_in[16];
    float* out = (float*)d_out;

    // ---- carve workspace (all offsets 256B aligned) ----
    char* wp = (char*)d_ws;
    auto take = [&](size_t bytes) {
        char* p = wp; wp += (bytes + 255) & ~(size_t)255; return (void*)p;
    };
    __bf16* hbf   = (__bf16*)take((size_t)TT * Dq * 2);       // rmsnorm1 out, bf16
    __bf16* wbf   = (__bf16*)take((size_t)4096 * 1024 * 2);   // shared transposed-weight buf
    float*  xz    = (float*) take((size_t)TT * 2 * DI * 4);   // in_proj out
    float*  xcs   = (float*) take((size_t)TT * DI * 4);       // conv+silu out
    float*  proj  = (float*) take((size_t)TT * NPROJ * 4);    // x_proj out
    float*  dts   = (float*) take((size_t)TT * DI * 4);       // softplus(dt)
    __bf16* ybf   = (__bf16*)take((size_t)TT * DI * 2);       // gated scan out, bf16
    __bf16* htbf  = (__bf16*)take((size_t)TT * Dq * 2);       // rmsnorm2 out, bf16
    float*  htf   = (float*) take((size_t)TT * Dq * 4);       // rmsnorm2 out, f32
    float*  gates = (float*) take((size_t)TT * NE * 4);       // dense gate matrix
    float*  stats = (float*) take(32 * 4);                    // psum[8] | dsum[8]
    __bf16* hidbf = (__bf16*)take((size_t)TT * DI * 2);       // expert hidden, bf16

    // ---------------- Mamba branch ----------------
    rmsnorm_k<<<TT, 256, 0, stream>>>(x, n1w, hbf, nullptr);
    wtrans_k<<<dim3((2 * DI) / 32, Dq / 32), 256, 0, stream>>>(inpw, wbf, Dq, 2 * DI);
    gemm_bf16_wmma<<<dim3((2 * DI) / 128, TT / 128), 256, 0, stream>>>(
        hbf, wbf, TT, 2 * DI, Dq, xz, nullptr, nullptr, nullptr, 0, nullptr, 0);
    conv_silu_k<<<(TT * DI) / 256, 256, 0, stream>>>(xz, convw, convb, xcs);
    xproj_k<<<(TT * NPROJ) / 256, 256, 0, stream>>>(xcs, xpw, proj);
    dt_k<<<(TT * DI) / 256, 256, 0, stream>>>(proj, dtw, dtb, dts);
    scan_k<<<(Bq * DI) / 256, 256, 0, stream>>>(xcs, dts, proj, xz, alog, dpar, ybf);
    wtrans_k<<<dim3(Dq / 32, DI / 32), 256, 0, stream>>>(outw, wbf, DI, Dq);
    gemm_bf16_wmma<<<dim3(Dq / 128, TT / 128), 256, 0, stream>>>(
        ybf, wbf, TT, Dq, DI, out, x, nullptr, nullptr, 0, nullptr, 1); // out = x + y@Wo

    // ---------------- MoE branch ----------------
    rmsnorm_k<<<TT, 256, 0, stream>>>(out, n2w, htbf, htf);
    zero32_k<<<1, 32, 0, stream>>>(stats);
    router_k<<<TT / 8, 256, 0, stream>>>(htf, rw, gates, stats, stats + 16);
    aux_k<<<1, 1, 0, stream>>>(stats, stats + 16, out + (size_t)TT * Dq);
    for (int e = 0; e < NE; ++e) {
        wtrans_k<<<dim3(DI / 32, Dq / 32), 256, 0, stream>>>(
            w1 + (size_t)e * Dq * DI, wbf, Dq, DI);
        gemm_bf16_wmma<<<dim3(DI / 128, TT / 128), 256, 0, stream>>>(
            htbf, wbf, TT, DI, Dq, nullptr, nullptr, b1 + (size_t)e * DI,
            nullptr, 0, hidbf, 2);                            // hid = gelu(.+b1)
        wtrans_k<<<dim3(Dq / 32, DI / 32), 256, 0, stream>>>(
            w2 + (size_t)e * DI * Dq, wbf, DI, Dq);
        gemm_bf16_wmma<<<dim3(Dq / 128, TT / 128), 256, 0, stream>>>(
            hidbf, wbf, TT, Dq, DI, out, nullptr, b2 + (size_t)e * Dq,
            gates, e, nullptr, 3);                            // out += g*(.+b2)
    }
}
